// TabNetEncoder_13950053777431
// MI455X (gfx1250) — matrix operand order repaired and checked by reading, not compiled, not run on hardware
//
#include <hip/hip_runtime.h>
#include <cstdint>
#include <cstddef>

typedef __attribute__((ext_vector_type(16))) __bf16 v16bf;
typedef __attribute__((ext_vector_type(8)))  float  v8f;
typedef unsigned short u16;
typedef unsigned int   u32;
typedef __attribute__((ext_vector_type(4)))  u32    u32x4;
typedef __attribute__((ext_vector_type(4)))  float  f32x4;

#define BSZ    262144
#define NFEAT  256
#define VBS    1024
#define NCHUNK (BSZ / VBS)
#define GAMMA_C 1.3f
#define EPS_BN  1e-5f
#define SQRT_HALF_C 0.7071067811865476f
#define SL_SCALE (1.0f / (262144.0f * 256.0f * 3.0f))

__device__ __forceinline__ float bf2f(u16 h) {
  union { u32 u; float f; } c; c.u = ((u32)h) << 16; return c.f;
}
__device__ __forceinline__ u16 f2bf(float f) {
  union { float f; u32 u; } c; c.f = f;
  u32 r = c.u + 0x7fffu + ((c.u >> 16) & 1u);
  return (u16)(r >> 16);
}
__device__ __forceinline__ float wred(float v) {
  #pragma unroll
  for (int o = 16; o > 0; o >>= 1) v += __shfl_xor(v, o, 32);
  return v;
}

union Frag { v16bf v; u32x4 q[2]; };

// ---------------------------------------------------------------- init / BN
__global__ __launch_bounds__(256) void k_init(float* dsl, float* wsf) {
  size_t i = (size_t)blockIdx.x * 256 + threadIdx.x;
  if (i < (size_t)BSZ * 64 + 1) dsl[i] = 0.f;
  if (i < 512) wsf[i] = 0.f;
}

__global__ __launch_bounds__(256) void k_bnsum(const float* __restrict__ x,
                                               float* __restrict__ s,
                                               float* __restrict__ q) {
  int c = threadIdx.x;
  size_t rb = (size_t)blockIdx.x * 256;
  float ls = 0.f, lq = 0.f;
  for (int r = 0; r < 256; ++r) {
    float v = x[(rb + r) * NFEAT + c];
    ls += v; lq += v * v;
  }
  atomicAdd(&s[c], ls);
  atomicAdd(&q[c], lq);
}

__global__ __launch_bounds__(256) void k_bnfin(const float* __restrict__ s,
                                               const float* __restrict__ q,
                                               const float* __restrict__ g,
                                               const float* __restrict__ b,
                                               float* __restrict__ sc,
                                               float* __restrict__ bi) {
  int c = threadIdx.x;
  float mu  = s[c] * (1.f / (float)BSZ);
  float var = q[c] * (1.f / (float)BSZ) - mu * mu;
  float A = rsqrtf(var + EPS_BN) * g[c];
  sc[c] = A;
  bi[c] = b[c] - mu * A;
}

// ------------------------------------------- attention matmul + ghost-BN
// Grid: (NCHUNK, 4). Block: 1024 (32 waves, 32 rows/wave). A = h_a
// (1024x64 bf16, cols 64..127 of hB, row stride 128). Out: 64-col tile of a.
__global__ __launch_bounds__(1024) void k_att(const u16* __restrict__ hb,
                                              const float* __restrict__ Watt,
                                              const float* __restrict__ g,
                                              const float* __restrict__ b,
                                              u16* __restrict__ aout, int zeroA) {
  const int chunk = blockIdx.x, cb = blockIdx.y;
  const int tid = threadIdx.x;
  if (zeroA) {  // step 0: h_a == 0 -> ghost_bn(0) = b_att
    for (int i = 0; i < 64; ++i) {
      int e = tid + (i << 10);
      int row = e >> 6, c = e & 63;
      int colg = cb * 64 + c;
      aout[(size_t)(chunk * VBS + row) * NFEAT + colg] = f2bf(b[colg]);
    }
    return;
  }
  __shared__ u16 As[VBS * 40];        // 1024 rows x 32-K slab, stride 40 (16B aligned)
  __shared__ u16 Wt[64 * 72];         // transposed Wt[n][k], stride 72 (16B aligned)
  __shared__ float sred[2][64];
  const int w = tid >> 5, l = tid & 31, half = l >> 4, ml = l & 15;
  if (tid < 128) ((float*)sred)[tid] = 0.f;
  // W_att tile -> LDS transposed (lane=N layout for B fragments), float4 reads
  {
    int ng = tid & 15, k = tid >> 4;
    f32x4 wv = *(const f32x4*)(Watt + (size_t)k * NFEAT + cb * 64 + ng * 4);
    #pragma unroll
    for (int j2 = 0; j2 < 4; ++j2) Wt[(ng * 4 + j2) * 72 + k] = f2bf(wv[j2]);
  }
  v8f acc[2][4];
  v8f zz = {0, 0, 0, 0, 0, 0, 0, 0};
  #pragma unroll
  for (int rt = 0; rt < 2; ++rt)
    #pragma unroll
    for (int ct = 0; ct < 4; ++ct) acc[rt][ct] = zz;
  const u32* hbw = (const u32*)hb;
  for (int kt = 0; kt < 2; ++kt) {
    __syncthreads();
    #pragma unroll
    for (int i = 0; i < 4; ++i) {   // b128 global -> b128 LDS
      int j = tid + (i << 10);
      int row = j >> 2, q = j & 3;
      const u32* src =
          hbw + (size_t)(chunk * VBS + row) * 64 + 32 + kt * 16 + q * 4;
      *(u32x4*)((u32*)As + row * 20 + q * 4) = *(const u32x4*)src;
    }
    __syncthreads();
    Frag af[2];
    #pragma unroll
    for (int rt = 0; rt < 2; ++rt) {
      const u32* arow = (const u32*)As + (w * 32 + rt * 16 + ml) * 20 + half * 4;
      af[rt].q[0] = *(const u32x4*)(arow);
      af[rt].q[1] = *(const u32x4*)(arow + 8);
    }
    #pragma unroll
    for (int ct = 0; ct < 4; ++ct) {
      Frag bf;
      const u32* wrow =
          (const u32*)Wt + (ct * 16 + ml) * 36 + kt * 16 + half * 4;
      bf.q[0] = *(const u32x4*)(wrow);
      bf.q[1] = *(const u32x4*)(wrow + 8);
      #pragma unroll
      for (int rt = 0; rt < 2; ++rt)
        acc[rt][ct] = __builtin_amdgcn_wmma_f32_16x16x32_bf16(
            false, af[rt].v, false, bf.v, (short)0, acc[rt][ct], false, false);
    }
  }
  // ghost-BN column stats over the 1024 rows held in this workgroup
  #pragma unroll
  for (int ct = 0; ct < 4; ++ct) {
    float s = 0.f, q = 0.f;
    #pragma unroll
    for (int rt = 0; rt < 2; ++rt)
      #pragma unroll
      for (int r = 0; r < 8; ++r) { float v = acc[rt][ct][r]; s += v; q += v * v; }
    atomicAdd(&sred[0][ct * 16 + ml], s);
    atomicAdd(&sred[1][ct * 16 + ml], q);
  }
  __syncthreads();
  #pragma unroll
  for (int ct = 0; ct < 4; ++ct) {
    int lc = ct * 16 + ml, colg = cb * 64 + lc;
    float mu  = sred[0][lc] * (1.f / (float)VBS);
    float var = sred[1][lc] * (1.f / (float)VBS) - mu * mu;
    float A = rsqrtf(var + EPS_BN) * g[colg];
    float C = b[colg] - mu * A;
    #pragma unroll
    for (int rt = 0; rt < 2; ++rt)
      #pragma unroll
      for (int r = 0; r < 8; ++r) {
        int rowl = w * 32 + rt * 16 + r + half * 8;
        aout[(size_t)(chunk * VBS + rowl) * NFEAT + colg] =
            f2bf(acc[rt][ct][r] * A + C);
      }
  }
}

// -------------------- sparsemax + prior update + mask/entropy + M*x_bn
// One wave per row (8 feats/lane). Fixed-point iteration on tau (exact
// sparsemax threshold: support shrinks monotonically to the true support).
__global__ __launch_bounds__(256) void k_mask(const u16* __restrict__ a,
                                              u16* __restrict__ prior,
                                              const float* __restrict__ x,
                                              const float* __restrict__ bnsc,
                                              const float* __restrict__ bnbi,
                                              float* __restrict__ masks,
                                              u16* __restrict__ mx,
                                              float* __restrict__ sl,
                                              int first, int last) {
  __shared__ float blk;
  if (threadIdx.x == 0) blk = 0.f;
  __syncthreads();
  const int w = threadIdx.x >> 5, l = threadIdx.x & 31;
  const size_t row = (size_t)blockIdx.x * 8 + w;
  float z[8], p[8];
  #pragma unroll
  for (int i = 0; i < 8; ++i) {
    int f = i * 32 + l;
    float pv = first ? 1.0f : bf2f(prior[row * NFEAT + f]);
    p[i] = pv;
    z[i] = pv * bf2f(a[row * NFEAT + f]);
  }
  float tau = -1e30f;
  for (int it = 0; it < 16; ++it) {
    float s = 0.f, k = 0.f;
    #pragma unroll
    for (int i = 0; i < 8; ++i)
      if (z[i] > tau) { s += z[i]; k += 1.f; }
    s = wred(s); k = wred(k);
    tau = (s - 1.f) / fmaxf(k, 1.f);
  }
  float sll = 0.f;
  #pragma unroll
  for (int i = 0; i < 8; ++i) {
    int f = i * 32 + l;
    float M = fmaxf(z[i] - tau, 0.f);
    masks[row * NFEAT + f] = M;
    if (!last) prior[row * NFEAT + f] = f2bf(p[i] * (GAMMA_C - M));
    float xb = x[row * NFEAT + f] * bnsc[f] + bnbi[f];
    mx[row * NFEAT + f] = f2bf(M * xb);
    sll += -M * __logf(M + 1e-15f);
  }
  sll = wred(sll);
  if (l == 0) atomicAdd(&blk, sll);
  __syncthreads();
  if (threadIdx.x == 0) atomicAdd(sl, blk * SL_SCALE);
}

// -------- fused GLU block: Y = in @ W ; ghost-BN(Y) ; h1*sigmoid(h2) [+res]
// Grid: (NCHUNK, 4 = 32-col output tile). Block 1024 = 32 waves, 32 rows/wave.
// Kdim in {128,256}; A streamed through a 32-wide K slab in LDS; W tile stored
// transposed (lane=N) for B fragments. Whole 1024x32-pair matmul output lives
// in accumulators (8 v8f/lane) -> single-pass ghost-BN via ds_add_f32.
__global__ __launch_bounds__(1024) void k_glu(const u16* __restrict__ in, int Kdim,
                                              const float* __restrict__ W,
                                              const float* __restrict__ g,
                                              const float* __restrict__ b,
                                              u16* __restrict__ out, int residual) {
  __shared__ u16 As[VBS * 40];       // 81920 B, row stride 40 (16B aligned)
  __shared__ u16 Wt[2 * 32 * 136];   // 17408 B : Wt[p][n][k], stride 136
  __shared__ float sred[2][2][32];   // [sum|sq][p][col]
  const int chunk = blockIdx.x, c0 = blockIdx.y;
  const int tid = threadIdx.x;
  const int w = tid >> 5, l = tid & 31, half = l >> 4, ml = l & 15;
  if (tid < 128) ((float*)sred)[tid] = 0.f;
  v8f acc[2][2][2];
  v8f zz = {0, 0, 0, 0, 0, 0, 0, 0};
  #pragma unroll
  for (int rt = 0; rt < 2; ++rt)
    #pragma unroll
    for (int p = 0; p < 2; ++p)
      #pragma unroll
      for (int ct = 0; ct < 2; ++ct) acc[rt][p][ct] = zz;
  const u32* inw = (const u32*)in;
  const int strideDw = Kdim >> 1;
  const int nhalf = Kdim >> 7;
  for (int kh = 0; kh < nhalf; ++kh) {
    __syncthreads();
    #pragma unroll
    for (int i = 0; i < 2; ++i) {   // W tile, float4 reads
      int e = tid + (i << 10);
      int ng = e & 7, k = (e >> 3) & 127, p = e >> 10;
      f32x4 wv = *(const f32x4*)(W + (size_t)(kh * 128 + k) * 256 + p * 128 +
                                 c0 * 32 + ng * 4);
      #pragma unroll
      for (int j2 = 0; j2 < 4; ++j2)
        Wt[(p * 32 + ng * 4 + j2) * 136 + k] = f2bf(wv[j2]);
    }
    for (int kt = 0; kt < 4; ++kt) {
      __syncthreads();
      #pragma unroll
      for (int i = 0; i < 4; ++i) {  // b128 global -> b128 LDS
        int j = tid + (i << 10);
        int row = j >> 2, q = j & 3;
        const u32* src = inw + (size_t)(chunk * VBS + row) * strideDw + kh * 64 +
                         kt * 16 + q * 4;
        *(u32x4*)((u32*)As + row * 20 + q * 4) = *(const u32x4*)src;
      }
      __syncthreads();
      Frag af[2];
      #pragma unroll
      for (int rt = 0; rt < 2; ++rt) {
        const u32* arow = (const u32*)As + (w * 32 + rt * 16 + ml) * 20 + half * 4;
        af[rt].q[0] = *(const u32x4*)(arow);
        af[rt].q[1] = *(const u32x4*)(arow + 8);
      }
      #pragma unroll
      for (int p = 0; p < 2; ++p)
        #pragma unroll
        for (int ct = 0; ct < 2; ++ct) {
          Frag bf;
          const u32* wrow =
              (const u32*)Wt + (p * 32 + ct * 16 + ml) * 68 + kt * 16 + half * 4;
          bf.q[0] = *(const u32x4*)(wrow);
          bf.q[1] = *(const u32x4*)(wrow + 8);
          #pragma unroll
          for (int rt = 0; rt < 2; ++rt)
            acc[rt][p][ct] = __builtin_amdgcn_wmma_f32_16x16x32_bf16(
                false, af[rt].v, false, bf.v, (short)0, acc[rt][p][ct], false,
                false);
        }
    }
  }
  // ghost-BN stats (full VBS rows resident in this workgroup's accumulators)
  #pragma unroll
  for (int p = 0; p < 2; ++p)
    #pragma unroll
    for (int ct = 0; ct < 2; ++ct) {
      float s = 0.f, q = 0.f;
      #pragma unroll
      for (int rt = 0; rt < 2; ++rt)
        #pragma unroll
        for (int r = 0; r < 8; ++r) {
          float v = acc[rt][p][ct][r];
          s += v; q += v * v;
        }
      atomicAdd(&sred[0][p][ct * 16 + ml], s);
      atomicAdd(&sred[1][p][ct * 16 + ml], q);
    }
  __syncthreads();
  #pragma unroll
  for (int ct = 0; ct < 2; ++ct) {
    int lc = ct * 16 + ml;
    int col = c0 * 32 + lc;
    float mu1 = sred[0][0][lc] * (1.f / (float)VBS);
    float v1  = sred[1][0][lc] * (1.f / (float)VBS) - mu1 * mu1;
    float mu2 = sred[0][1][lc] * (1.f / (float)VBS);
    float v2  = sred[1][1][lc] * (1.f / (float)VBS) - mu2 * mu2;
    float A1 = rsqrtf(v1 + EPS_BN) * g[col];       float C1 = b[col] - mu1 * A1;
    float A2 = rsqrtf(v2 + EPS_BN) * g[128 + col]; float C2 = b[128 + col] - mu2 * A2;
    #pragma unroll
    for (int rt = 0; rt < 2; ++rt)
      #pragma unroll
      for (int r = 0; r < 8; ++r) {
        int rowl = w * 32 + rt * 16 + r + half * 8;
        size_t grow = (size_t)(chunk * VBS + rowl);
        float y1 = acc[rt][0][ct][r] * A1 + C1;
        float y2 = acc[rt][1][ct][r] * A2 + C2;
        float o = y1 * (1.f / (1.f + __expf(-y2)));
        if (residual) o = (bf2f(in[grow * 128 + col]) + o) * SQRT_HALF_C;
        out[grow * 128 + col] = f2bf(o);
      }
  }
}

// ------------------------------------------------ d_out += relu(h[:, :64])
__global__ __launch_bounds__(256) void k_epi(const u16* __restrict__ h,
                                             float* __restrict__ d) {
  size_t i = ((size_t)blockIdx.x * 256 + threadIdx.x) * 4;
  size_t row = i >> 6;
  int col = (int)(i & 63);
  const u32* hw = (const u32*)(h + row * 128 + col);
  u32 h01 = hw[0], h23 = hw[1];
  f32x4 dv = *(f32x4*)(d + i);
  dv[0] += fmaxf(bf2f((u16)(h01 & 0xffffu)), 0.f);
  dv[1] += fmaxf(bf2f((u16)(h01 >> 16)), 0.f);
  dv[2] += fmaxf(bf2f((u16)(h23 & 0xffffu)), 0.f);
  dv[3] += fmaxf(bf2f((u16)(h23 >> 16)), 0.f);
  *(f32x4*)(d + i) = dv;
}

// ---------------------------------------------------------------- launcher
extern "C" void kernel_launch(void* const* d_in, const int* in_sizes, int n_in,
                              void* d_out, int out_size, void* d_ws, size_t ws_size,
                              hipStream_t stream) {
  (void)in_sizes; (void)n_in; (void)out_size; (void)ws_size;
  const float* x      = (const float*)d_in[0];
  const float* bn0_g  = (const float*)d_in[1];
  const float* bn0_b  = (const float*)d_in[2];
  const float* W_sh0  = (const float*)d_in[3];
  const float* g_sh0  = (const float*)d_in[4];
  const float* b_sh0  = (const float*)d_in[5];
  const float* W_sh1  = (const float*)d_in[6];
  const float* g_sh1  = (const float*)d_in[7];
  const float* b_sh1  = (const float*)d_in[8];
  const float* W_step = (const float*)d_in[9];
  const float* g_step = (const float*)d_in[10];
  const float* b_step = (const float*)d_in[11];
  const float* W_att  = (const float*)d_in[12];
  const float* g_att  = (const float*)d_in[13];
  const float* b_att  = (const float*)d_in[14];

  float* dout  = (float*)d_out;
  float* sl    = dout + (size_t)BSZ * 64;
  float* masks = sl + 1;

  char* wsb = (char*)d_ws;
  float* bn_sum = (float*)wsb;
  float* bn_sq  = bn_sum + 256;
  float* bn_sc  = bn_sum + 512;
  float* bn_bi  = bn_sum + 768;
  u16* prior = (u16*)(wsb + 4096);
  u16* abuf  = prior + (size_t)BSZ * NFEAT;
  u16* mx    = abuf  + (size_t)BSZ * NFEAT;
  u16* hA    = mx    + (size_t)BSZ * NFEAT;
  u16* hB    = hA    + (size_t)BSZ * 128;

  k_init<<<65537, 256, 0, stream>>>(dout, bn_sum);
  k_bnsum<<<1024, 256, 0, stream>>>(x, bn_sum, bn_sq);
  k_bnfin<<<1, 256, 0, stream>>>(bn_sum, bn_sq, bn0_g, bn0_b, bn_sc, bn_bi);

  for (int s = 0; s < 3; ++s) {
    k_att<<<dim3(NCHUNK, 4), 1024, 0, stream>>>(
        hB, W_att + (size_t)s * 64 * NFEAT, g_att + s * NFEAT, b_att + s * NFEAT,
        abuf, s == 0 ? 1 : 0);
    k_mask<<<BSZ / 8, 256, 0, stream>>>(abuf, prior, x, bn_sc, bn_bi,
                                        masks + (size_t)s * BSZ * NFEAT, mx, sl,
                                        s == 0 ? 1 : 0, s == 2 ? 1 : 0);
    k_glu<<<dim3(NCHUNK, 4), 1024, 0, stream>>>(mx, 256, W_sh0, g_sh0, b_sh0, hA, 0);
    k_glu<<<dim3(NCHUNK, 4), 1024, 0, stream>>>(hA, 128, W_sh1, g_sh1, b_sh1, hB, 1);
    k_glu<<<dim3(NCHUNK, 4), 1024, 0, stream>>>(
        hB, 128, W_step + (size_t)(s * 2 + 0) * 128 * NFEAT,
        g_step + (s * 2 + 0) * NFEAT, b_step + (s * 2 + 0) * NFEAT, hA, 1);
    k_glu<<<dim3(NCHUNK, 4), 1024, 0, stream>>>(
        hA, 128, W_step + (size_t)(s * 2 + 1) * 128 * NFEAT,
        g_step + (s * 2 + 1) * NFEAT, b_step + (s * 2 + 1) * NFEAT, hB, 1);
    k_epi<<<BSZ * 64 / 1024, 256, 0, stream>>>(hB, dout);
  }
}
